// HybridDetector_78554951843997
// MI455X (gfx1250) — matrix operand order repaired and checked by reading, not compiled, not run on hardware
//
#include <hip/hip_runtime.h>
#include <math.h>

// ---------------------------------------------------------------------------
// CDNA5 (gfx1250) wave32 WMMA types / helpers
// ---------------------------------------------------------------------------
typedef __attribute__((ext_vector_type(16))) __bf16 v16bf;
typedef __attribute__((ext_vector_type(8)))  float  v8f;
typedef __attribute__((ext_vector_type(4)))  unsigned int v4u;
typedef __attribute__((ext_vector_type(8)))  unsigned int v8u;

__device__ __forceinline__ float bf2f(unsigned short h) {
  unsigned int u = ((unsigned int)h) << 16;
  return __builtin_bit_cast(float, u);
}
__device__ __forceinline__ unsigned short f2bf(float f) {
  unsigned int u = __builtin_bit_cast(unsigned int, f);
  unsigned int r = u + 0x7FFFu + ((u >> 16) & 1u);  // round-to-nearest-even
  return (unsigned short)(r >> 16);
}
// Assemble a 16-element bf16 fragment from two 16-byte loads (register coalesce)
__device__ __forceinline__ v16bf pack2(v4u lo, v4u hi) {
  v8u t;
  t[0] = lo[0]; t[1] = lo[1]; t[2] = lo[2]; t[3] = lo[3];
  t[4] = hi[0]; t[5] = hi[1]; t[6] = hi[2]; t[7] = hi[3];
  return __builtin_bit_cast(v16bf, t);
}
__device__ __forceinline__ v4u ld128(const unsigned short* p) {
  return *(const v4u*)p;
}

// ---------------------------------------------------------------------------
// Implicit-GEMM 3x3 conv, NHWC bf16 activations, weights [tap][Cout][Cin] bf16.
// One wave = 32(pixel) x 16(Cout) tile: two v_wmma_f32_16x16x32_bf16 per chunk
// sharing the B (weight) fragment. Out-of-bounds taps read a zeroed row
// (branchless base-pointer select, no EXEC churn in the hot loop).
// mode: 0 = bias only, 1 = relu(conv+bias), 2 = res + relu(conv+bias)
// ---------------------------------------------------------------------------
__global__ __launch_bounds__(256) void conv3x3_wmma_kernel(
    const unsigned short* __restrict__ in, const unsigned short* __restrict__ wt,
    const float* __restrict__ bias, const unsigned short* __restrict__ res,
    unsigned short* __restrict__ out, const unsigned short* __restrict__ zrow,
    int Bn, int H, int W, int Cin, int Cout, int stride, int mode,
    int tilesN, int totalTiles) {
  int wave = (int)((blockIdx.x * blockDim.x + threadIdx.x) >> 5);
  if (wave >= totalTiles) return;  // uniform per wave; EXEC all-1s inside
  int lane = threadIdx.x & 31, half = lane >> 4, lr = lane & 15;
  int tp = wave / tilesN, tn = wave % tilesN;
  int Ho = (stride == 1) ? H : (H >> 1);
  int Wo = (stride == 1) ? W : (W >> 1);
  int hw = Ho * Wo;
  // two 16-pixel tiles per wave
  int p0 = tp * 32 + lr, p1 = p0 + 16;
  int b0 = p0 / hw, r0 = p0 - b0 * hw, oy0 = r0 / Wo, ox0 = r0 - oy0 * Wo;
  int b1 = p1 / hw, r1 = p1 - b1 * hw, oy1 = r1 / Wo, ox1 = r1 - oy1 * Wo;
  int padlo = (stride == 1) ? 1 : 0;  // JAX 'SAME': s=1 -> lo=1, s=2 -> lo=0
  int iyb0 = oy0 * stride - padlo, ixb0 = ox0 * stride - padlo;
  int iyb1 = oy1 * stride - padlo, ixb1 = ox1 * stride - padlo;
  int nbase = tn * 16;
  v8f acc0 = {0.f, 0.f, 0.f, 0.f, 0.f, 0.f, 0.f, 0.f};
  v8f acc1 = acc0;
  int ha = half << 3;   // A: K offset of this lane-half within a chunk
  int hb = half << 3;   // B: K offset of this lane-half within a chunk
  for (int ky = 0; ky < 3; ky++) {
    for (int kx = 0; kx < 3; kx++) {
      int iy0 = iyb0 + ky, ix0 = ixb0 + kx;
      int iy1 = iyb1 + ky, ix1 = ixb1 + kx;
      bool ok0 = (iy0 >= 0) & (iy0 < H) & (ix0 >= 0) & (ix0 < W);
      bool ok1 = (iy1 >= 0) & (iy1 < H) & (ix1 >= 0) & (ix1 < W);
      const unsigned short* a0 =
          ok0 ? in + ((long long)(b0 * H + iy0) * W + ix0) * Cin : zrow;
      const unsigned short* a1 =
          ok1 ? in + ((long long)(b1 * H + iy1) * W + ix1) * Cin : zrow;
      const unsigned short* wl =
          wt + ((size_t)(ky * 3 + kx) * Cout + nbase + lr) * (size_t)Cin;
      for (int cb = 0; cb < Cin; cb += 32) {
        v16bf bfm = pack2(ld128(wl + cb + hb), ld128(wl + cb + hb + 16));
        v16bf af0 = pack2(ld128(a0 + cb + ha), ld128(a0 + cb + ha + 16));
        v16bf af1 = pack2(ld128(a1 + cb + ha), ld128(a1 + cb + ha + 16));
        acc0 = __builtin_amdgcn_wmma_f32_16x16x32_bf16(false, af0, false, bfm,
                                                       (short)0, acc0, false, false);
        acc1 = __builtin_amdgcn_wmma_f32_16x16x32_bf16(false, af1, false, bfm,
                                                       (short)0, acc1, false, false);
      }
    }
  }
  int n = nbase + lr;
  float bv = bias ? bias[n] : 0.f;
#pragma unroll
  for (int v = 0; v < 8; v++) {
    int m = v + ha;  // C/D: lane holds column n, VGPR v holds row m
    size_t oi0 = (size_t)(tp * 32 + m) * (size_t)Cout + n;
    size_t oi1 = (size_t)(tp * 32 + 16 + m) * (size_t)Cout + n;
    float v0 = acc0[v] + bv, v1 = acc1[v] + bv;
    if (mode >= 1) { v0 = fmaxf(v0, 0.f); v1 = fmaxf(v1, 0.f); }
    if (mode == 2) { v0 += bf2f(res[oi0]); v1 += bf2f(res[oi1]); }
    out[oi0] = f2bf(v0);
    out[oi1] = f2bf(v1);
  }
}

// ---------------------------------------------------------------------------
// GEMM: out[M,N] = act(A[M,K] * Bt[N,K]^T + bias) (+ resF). A,Bt bf16.
// One wave = 32(M) x 16(N); act: 0 none, 1 relu, 2 gelu(tanh).
// ---------------------------------------------------------------------------
__global__ __launch_bounds__(256) void gemm_wmma_kernel(
    const unsigned short* __restrict__ A, const unsigned short* __restrict__ Bt,
    const float* __restrict__ bias, const float* __restrict__ resF,
    float* __restrict__ outF, unsigned short* __restrict__ outH,
    int M, int K, int N, int act, int tilesN, int totalTiles) {
  int wave = (int)((blockIdx.x * blockDim.x + threadIdx.x) >> 5);
  if (wave >= totalTiles) return;
  int lane = threadIdx.x & 31, half = lane >> 4, lr = lane & 15;
  int tp = wave / tilesN, tn = wave % tilesN;
  const unsigned short* arow0 = A + (size_t)(tp * 32 + lr) * (size_t)K;
  const unsigned short* arow1 = arow0 + (size_t)16 * K;
  const unsigned short* brow = Bt + (size_t)(tn * 16 + lr) * (size_t)K;
  v8f acc0 = {0.f, 0.f, 0.f, 0.f, 0.f, 0.f, 0.f, 0.f};
  v8f acc1 = acc0;
  int ha = half << 3, hb = half << 3;
  for (int cb = 0; cb < K; cb += 32) {
    v16bf bfm = pack2(ld128(brow + cb + hb), ld128(brow + cb + hb + 16));
    v16bf af0 = pack2(ld128(arow0 + cb + ha), ld128(arow0 + cb + ha + 16));
    v16bf af1 = pack2(ld128(arow1 + cb + ha), ld128(arow1 + cb + ha + 16));
    acc0 = __builtin_amdgcn_wmma_f32_16x16x32_bf16(false, af0, false, bfm,
                                                   (short)0, acc0, false, false);
    acc1 = __builtin_amdgcn_wmma_f32_16x16x32_bf16(false, af1, false, bfm,
                                                   (short)0, acc1, false, false);
  }
  int n = tn * 16 + lr;
  float bv = bias ? bias[n] : 0.f;
#pragma unroll
  for (int v = 0; v < 8; v++) {
    int m = v + ha;
    size_t oi0 = (size_t)(tp * 32 + m) * (size_t)N + n;
    size_t oi1 = (size_t)(tp * 32 + 16 + m) * (size_t)N + n;
    float v0 = acc0[v] + bv, v1 = acc1[v] + bv;
    if (act == 1) {
      v0 = fmaxf(v0, 0.f); v1 = fmaxf(v1, 0.f);
    } else if (act == 2) {
      float u0 = v0, u1 = v1;
      v0 = 0.5f * u0 * (1.f + tanhf(0.7978845608f * (u0 + 0.044715f * u0 * u0 * u0)));
      v1 = 0.5f * u1 * (1.f + tanhf(0.7978845608f * (u1 + 0.044715f * u1 * u1 * u1)));
    }
    if (resF) { v0 += resF[oi0]; v1 += resF[oi1]; }
    if (outF) { outF[oi0] = v0; outF[oi1] = v1; }
    else      { outH[oi0] = f2bf(v0); outH[oi1] = f2bf(v1); }
  }
}

// ---------------------------------------------------------------------------
// Stem: normalize + 3x3/s2 conv (3->32) + relu, NCHW f32 in -> NHWC bf16 out
// ---------------------------------------------------------------------------
__global__ void stem_kernel(const float* __restrict__ x, const float* __restrict__ w,
                            const float* __restrict__ bias,
                            unsigned short* __restrict__ out) {
  const int H = 512, W = 512, Ho = 256, Wo = 256, CO = 32;
  int idx = blockIdx.x * blockDim.x + threadIdx.x;
  int total = 8 * Ho * Wo * CO;
  if (idx >= total) return;
  int co = idx % CO; int t = idx / CO;
  int ox = t % Wo; t /= Wo; int oy = t % Ho; int b = t / Ho;
  const float mean[3] = {0.485f, 0.456f, 0.406f};
  const float isd[3]  = {1.f / 0.229f, 1.f / 0.224f, 1.f / 0.225f};
  float acc = bias[co];
  for (int ci = 0; ci < 3; ci++) {
    for (int ky = 0; ky < 3; ky++) {
      int iy = oy * 2 + ky; if (iy >= H) continue;   // pad_lo = 0 for s2 SAME
      for (int kx = 0; kx < 3; kx++) {
        int ix = ox * 2 + kx; if (ix >= W) continue;
        float v = x[(((size_t)b * 3 + ci) * H + iy) * W + ix];
        v = (v - mean[ci]) * isd[ci];
        acc += v * w[((co * 3 + ci) * 3 + ky) * 3 + kx];
      }
    }
  }
  out[idx] = f2bf(fmaxf(acc, 0.f));
}

// ---------------------------------------------------------------------------
// LayerNorm over C per token: f32 in -> bf16 out (normalized * g + b)
// ---------------------------------------------------------------------------
__global__ void layernorm_kernel(const float* __restrict__ t,
                                 const float* __restrict__ g,
                                 const float* __restrict__ b,
                                 unsigned short* __restrict__ out,
                                 int total, int C) {
  int i = blockIdx.x * blockDim.x + threadIdx.x;
  if (i >= total) return;
  const float* row = t + (size_t)i * C;
  float s = 0.f;
  for (int c = 0; c < C; c++) s += row[c];
  float mean = s / C;
  float v = 0.f;
  for (int c = 0; c < C; c++) { float d = row[c] - mean; v += d * d; }
  float r = rsqrtf(v / C + 1e-5f);
  unsigned short* orow = out + (size_t)i * C;
  for (int c = 0; c < C; c++) orow[c] = f2bf((row[c] - mean) * r * g[c] + b[c]);
}

// ---------------------------------------------------------------------------
// Fused streaming attention (2-pass softmax). qkv bf16 [B,N,3C]; out bf16 [B,N,C]
// ---------------------------------------------------------------------------
template <int HD>
__global__ void attn_kernel(const unsigned short* __restrict__ qkv,
                            unsigned short* __restrict__ o,
                            int Bn, int Ntok, int C, int heads) {
  int idx = blockIdx.x * blockDim.x + threadIdx.x;
  int total = Bn * heads * Ntok;
  if (idx >= total) return;
  int n = idx % Ntok; int bh = idx / Ntok; int h = bh % heads; int b = bh / heads;
  const unsigned short* qp = qkv + ((size_t)(b * Ntok + n) * 3 * C) + h * HD;
  float q[HD];
#pragma unroll
  for (int d = 0; d < HD; d++) q[d] = bf2f(qp[d]);
  float scale = rsqrtf((float)HD);
  const unsigned short* base = qkv + (size_t)b * Ntok * 3 * C;
  float mx = -1e30f;
  for (int m = 0; m < Ntok; m++) {
    const unsigned short* kp = base + (size_t)m * 3 * C + C + h * HD;
    float s = 0.f;
#pragma unroll
    for (int d = 0; d < HD; d++) s += q[d] * bf2f(kp[d]);
    mx = fmaxf(mx, s * scale);
  }
  float accv[HD];
#pragma unroll
  for (int d = 0; d < HD; d++) accv[d] = 0.f;
  float l = 0.f;
  for (int m = 0; m < Ntok; m++) {
    const unsigned short* kp = base + (size_t)m * 3 * C + C + h * HD;
    float s = 0.f;
#pragma unroll
    for (int d = 0; d < HD; d++) s += q[d] * bf2f(kp[d]);
    float e = __expf(s * scale - mx);
    l += e;
    const unsigned short* vp = base + (size_t)m * 3 * C + 2 * C + h * HD;
#pragma unroll
    for (int d = 0; d < HD; d++) accv[d] += e * bf2f(vp[d]);
  }
  float rl = 1.f / l;
  unsigned short* op = o + ((size_t)(b * Ntok + n) * C) + h * HD;
#pragma unroll
  for (int d = 0; d < HD; d++) op[d] = f2bf(accv[d] * rl);
}

// ---------------------------------------------------------------------------
// Elementwise conversions & weight-prep kernels
// ---------------------------------------------------------------------------
__global__ void cvt_h2f_kernel(const unsigned short* __restrict__ a,
                               float* __restrict__ o, int n) {
  int i = blockIdx.x * blockDim.x + threadIdx.x;
  if (i < n) o[i] = bf2f(a[i]);
}
__global__ void cvt_f2h_kernel(const float* __restrict__ a,
                               unsigned short* __restrict__ o, int n) {
  int i = blockIdx.x * blockDim.x + threadIdx.x;
  if (i < n) o[i] = f2bf(a[i]);
}
__global__ void zero_us_kernel(unsigned short* __restrict__ p, int n) {
  int i = blockIdx.x * blockDim.x + threadIdx.x;
  if (i < n) p[i] = 0;
}
// OIHW f32 -> [tap][O][I] bf16
__global__ void convw_f2h_kernel(const float* __restrict__ w,
                                 unsigned short* __restrict__ o, int O, int I) {
  int i = blockIdx.x * blockDim.x + threadIdx.x;
  int total = 9 * O * I;
  if (i >= total) return;
  int ic = i % I; int t = i / I; int oc = t % O; int tap = t / O;
  o[i] = f2bf(w[(size_t)(oc * I + ic) * 9 + tap]);
}
// [K,N] f32 -> [N,K] bf16
__global__ void transpose_f2h_kernel(const float* __restrict__ w,
                                     unsigned short* __restrict__ o, int K, int N) {
  int i = blockIdx.x * blockDim.x + threadIdx.x;
  if (i >= K * N) return;
  int k = i % K; int n = i / K;
  o[(size_t)n * K + k] = f2bf(w[(size_t)k * N + n]);
}

// ---------------------------------------------------------------------------
// FPN top-down: lat[child] += nearest-up(parent), both bf16, in place on child
// ---------------------------------------------------------------------------
__global__ void add_up_kernel(unsigned short* __restrict__ lat,
                              const unsigned short* __restrict__ par,
                              int Bn, int Hc, int Wc, int C) {
  int i = blockIdx.x * blockDim.x + threadIdx.x;
  int total = Bn * Hc * Wc * C;
  if (i >= total) return;
  int c = i % C; int t = i / C;
  int x = t % Wc; t /= Wc; int y = t % Hc; int b = t / Hc;
  size_t pi = (((size_t)b * (Hc >> 1) + (y >> 1)) * (Wc >> 1) + (x >> 1)) * C + c;
  lat[i] = f2bf(bf2f(lat[i]) + bf2f(par[pi]));
}

// ---------------------------------------------------------------------------
// Head final: cls(10)/ctr(1)/box(4) 3x3 convs from towers, write [B,21760,15] f32
// ---------------------------------------------------------------------------
__global__ void head_final_kernel(
    const unsigned short* __restrict__ ct, const unsigned short* __restrict__ rt,
    const float* __restrict__ wcls, const float* __restrict__ bcls,
    const float* __restrict__ wctr, const float* __restrict__ bctr,
    const float* __restrict__ wbox, const float* __restrict__ bbox,
    float* __restrict__ out, int Bn, int Hh, int Wh, int Cin,
    int lvl_off, int total_tokens) {
  int idx = blockIdx.x * blockDim.x + threadIdx.x;
  int P = Bn * Hh * Wh;
  if (idx >= P * 15) return;
  int oc = idx % 15; int p = idx / 15;
  int hw = Hh * Wh;
  int b = p / hw; int rem = p - b * hw; int oy = rem / Wh; int ox = rem - oy * Wh;
  const unsigned short* src; const float* w; float acc; bool dorelu = false;
  if (oc < 10)      { src = ct; w = wcls + (size_t)oc * Cin * 9;        acc = bcls[oc]; }
  else if (oc == 10){ src = rt; w = wctr;                                acc = bctr[0]; }
  else              { src = rt; w = wbox + (size_t)(oc - 11) * Cin * 9;  acc = bbox[oc - 11]; dorelu = true; }
  for (int ky = 0; ky < 3; ky++) {
    int iy = oy - 1 + ky; if (iy < 0 || iy >= Hh) continue;
    for (int kx = 0; kx < 3; kx++) {
      int ix = ox - 1 + kx; if (ix < 0 || ix >= Wh) continue;
      const unsigned short* srow = src + (((size_t)(b * Hh + iy) * Wh + ix) * Cin);
      int tap = ky * 3 + kx;
      for (int ci = 0; ci < Cin; ci++) acc += bf2f(srow[ci]) * w[ci * 9 + tap];
    }
  }
  if (dorelu) acc = fmaxf(acc, 0.f);
  out[((size_t)b * total_tokens + lvl_off + rem) * 15 + oc] = acc;
}

// ---------------------------------------------------------------------------
// Host side
// ---------------------------------------------------------------------------
// d_in indices: jax tree_flatten order (nested dict keys sorted lexicographically)
enum {
  IN_BOX_B = 0, IN_BOX_W, IN_CLS_B, IN_CLS_W,
  IN_CT0_B, IN_CT0_W, IN_CT1_B, IN_CT1_W,
  IN_CTR_B, IN_CTR_W,
  IN_RT0_B, IN_RT0_W, IN_RT1_B, IN_RT1_W,
  IN_LAT2_B, IN_LAT2_W, IN_LAT3_B, IN_LAT3_W, IN_LAT4_B, IN_LAT4_W, IN_LAT5_B, IN_LAT5_W,
  IN_SM2_B, IN_SM2_W, IN_SM3_B, IN_SM3_W, IN_SM4_B, IN_SM4_W, IN_SM5_B, IN_SM5_W,
  IN_S0B0_B, IN_S0B0_W, IN_S0D_B, IN_S0D_W,
  IN_S1B0_B, IN_S1B0_W, IN_S1B1_B, IN_S1B1_W, IN_S1D_B, IN_S1D_W,
  IN_S2B0_B, IN_S2B0_W, IN_S2B1_B, IN_S2B1_W, IN_S2B2_B, IN_S2B2_W, IN_S2D_B, IN_S2D_W,
  IN_S3B0_B, IN_S3B0_W, IN_S3B1_B, IN_S3B1_W, IN_S3D_B, IN_S3D_W,
  IN_STEM_B, IN_STEM_W,
  IN_P4_LN1B, IN_P4_LN1G, IN_P4_LN2B, IN_P4_LN2G, IN_P4_MB1, IN_P4_MB2,
  IN_P4_MW1, IN_P4_MW2, IN_P4_PB, IN_P4_PW, IN_P4_QB, IN_P4_QW,
  IN_P5_LN1B, IN_P5_LN1G, IN_P5_LN2B, IN_P5_LN2G, IN_P5_MB1, IN_P5_MB2,
  IN_P5_MW1, IN_P5_MW2, IN_P5_PB, IN_P5_PW, IN_P5_QB, IN_P5_QW,
  IN_X
};

static inline unsigned gblk(long long n) { return (unsigned)((n + 255) / 256); }

extern "C" void kernel_launch(void* const* d_in, const int* in_sizes, int n_in,
                              void* d_out, int out_size, void* d_ws, size_t ws_size,
                              hipStream_t stream) {
  (void)in_sizes; (void)n_in; (void)out_size; (void)ws_size;
  char* wsbase = (char*)d_ws;
  size_t wsoff = 0;
  auto alloc = [&](size_t bytes) -> void* {
    wsoff = (wsoff + 255) & ~(size_t)255;
    void* p = wsbase + wsoff;
    wsoff += bytes;
    return p;
  };
  auto F = [&](int i) -> const float* { return (const float*)d_in[i]; };

  // zero row used for out-of-bounds conv taps (Cin max 256 + headroom)
  unsigned short* zrow = (unsigned short*)alloc(512 * 2);
  zero_us_kernel<<<2, 256, 0, stream>>>(zrow, 512);

  // ---- weight prep (bf16 conversions, recomputed deterministically per call)
  auto convw = [&](int i, int O, int I) -> unsigned short* {
    unsigned short* o = (unsigned short*)alloc((size_t)9 * O * I * 2);
    convw_f2h_kernel<<<gblk((long long)9 * O * I), 256, 0, stream>>>(F(i), o, O, I);
    return o;
  };
  auto matw = [&](int i, int K, int N) -> unsigned short* {
    unsigned short* o = (unsigned short*)alloc((size_t)K * N * 2);
    transpose_f2h_kernel<<<gblk((long long)K * N), 256, 0, stream>>>(F(i), o, K, N);
    return o;
  };
  auto cvt1 = [&](int i, int n) -> unsigned short* {  // 1x1 conv: already [O][I]
    unsigned short* o = (unsigned short*)alloc((size_t)n * 2);
    cvt_f2h_kernel<<<gblk(n), 256, 0, stream>>>(F(i), o, n);
    return o;
  };

  unsigned short* w_s0d  = convw(IN_S0D_W, 64, 32);
  unsigned short* w_s0b0 = convw(IN_S0B0_W, 64, 64);
  unsigned short* w_s1d  = convw(IN_S1D_W, 128, 64);
  unsigned short* w_s1b0 = convw(IN_S1B0_W, 128, 128);
  unsigned short* w_s1b1 = convw(IN_S1B1_W, 128, 128);
  unsigned short* w_s2d  = convw(IN_S2D_W, 192, 128);
  unsigned short* w_s2b0 = convw(IN_S2B0_W, 192, 192);
  unsigned short* w_s2b1 = convw(IN_S2B1_W, 192, 192);
  unsigned short* w_s2b2 = convw(IN_S2B2_W, 192, 192);
  unsigned short* w_s3d  = convw(IN_S3D_W, 256, 192);
  unsigned short* w_s3b0 = convw(IN_S3B0_W, 256, 256);
  unsigned short* w_s3b1 = convw(IN_S3B1_W, 256, 256);
  unsigned short* w_sm[4] = {convw(IN_SM2_W, 128, 128), convw(IN_SM3_W, 128, 128),
                             convw(IN_SM4_W, 128, 128), convw(IN_SM5_W, 128, 128)};
  unsigned short* w_ct0 = convw(IN_CT0_W, 128, 128);
  unsigned short* w_ct1 = convw(IN_CT1_W, 128, 128);
  unsigned short* w_rt0 = convw(IN_RT0_W, 128, 128);
  unsigned short* w_rt1 = convw(IN_RT1_W, 128, 128);
  unsigned short* w_lat[4] = {cvt1(IN_LAT2_W, 128 * 64), cvt1(IN_LAT3_W, 128 * 128),
                              cvt1(IN_LAT4_W, 128 * 192), cvt1(IN_LAT5_W, 128 * 256)};
  unsigned short* w_q4  = matw(IN_P4_QW, 192, 576);
  unsigned short* w_p4  = matw(IN_P4_PW, 192, 192);
  unsigned short* w_m14 = matw(IN_P4_MW1, 192, 768);
  unsigned short* w_m24 = matw(IN_P4_MW2, 768, 192);
  unsigned short* w_q5  = matw(IN_P5_QW, 256, 768);
  unsigned short* w_p5  = matw(IN_P5_PW, 256, 256);
  unsigned short* w_m15 = matw(IN_P5_MW1, 256, 1024);
  unsigned short* w_m25 = matw(IN_P5_MW2, 1024, 256);

  // ---- activation buffers (NHWC bf16 unless noted)
  unsigned short* A_stem = (unsigned short*)alloc((size_t)8 * 256 * 256 * 32 * 2);
  unsigned short* S1A = (unsigned short*)alloc((size_t)8 * 128 * 128 * 64 * 2);
  unsigned short* S1B = (unsigned short*)alloc((size_t)8 * 128 * 128 * 64 * 2);
  unsigned short* S2A = (unsigned short*)alloc((size_t)8 * 64 * 64 * 128 * 2);
  unsigned short* S2B = (unsigned short*)alloc((size_t)8 * 64 * 64 * 128 * 2);
  unsigned short* S3A = (unsigned short*)alloc((size_t)8 * 32 * 32 * 192 * 2);
  unsigned short* S3B = (unsigned short*)alloc((size_t)8 * 32 * 32 * 192 * 2);
  unsigned short* S4A = (unsigned short*)alloc((size_t)8 * 16 * 16 * 256 * 2);
  unsigned short* S4B = (unsigned short*)alloc((size_t)8 * 16 * 16 * 256 * 2);
  float*          Ttok = (float*)alloc((size_t)8 * 1024 * 192 * 4);      // residual (f32)
  unsigned short* Th   = (unsigned short*)alloc((size_t)8 * 1024 * 192 * 2);
  unsigned short* Tqkv = (unsigned short*)alloc((size_t)8 * 1024 * 576 * 2);
  unsigned short* Tatt = (unsigned short*)alloc((size_t)8 * 1024 * 192 * 2);
  unsigned short* Tmlp = (unsigned short*)alloc((size_t)8 * 1024 * 768 * 2);
  unsigned short* LAT[4] = {
      (unsigned short*)alloc((size_t)8 * 128 * 128 * 128 * 2),
      (unsigned short*)alloc((size_t)8 * 64 * 64 * 128 * 2),
      (unsigned short*)alloc((size_t)8 * 32 * 32 * 128 * 2),
      (unsigned short*)alloc((size_t)8 * 16 * 16 * 128 * 2)};
  unsigned short* PYR[4] = {
      (unsigned short*)alloc((size_t)8 * 128 * 128 * 128 * 2),
      (unsigned short*)alloc((size_t)8 * 64 * 64 * 128 * 2),
      (unsigned short*)alloc((size_t)8 * 32 * 32 * 128 * 2),
      (unsigned short*)alloc((size_t)8 * 16 * 16 * 128 * 2)};
  unsigned short* T1 = (unsigned short*)alloc((size_t)8 * 128 * 128 * 128 * 2);
  unsigned short* C2 = (unsigned short*)alloc((size_t)8 * 128 * 128 * 128 * 2);
  unsigned short* R2 = (unsigned short*)alloc((size_t)8 * 128 * 128 * 128 * 2);

  // ---- launch helpers
  auto conv = [&](const unsigned short* in, const unsigned short* wt, const float* bias,
                  const unsigned short* res, unsigned short* out,
                  int Bn, int H, int W, int Cin, int Cout, int stride, int mode) {
    int Ho = (stride == 1) ? H : H / 2, Wo = (stride == 1) ? W : W / 2;
    int tilesP = (Bn * Ho * Wo) / 32, tilesN = Cout / 16;  // 32 pixels per wave
    int total = tilesP * tilesN;
    conv3x3_wmma_kernel<<<(total + 7) / 8, 256, 0, stream>>>(
        in, wt, bias, res, out, zrow, Bn, H, W, Cin, Cout, stride, mode, tilesN, total);
  };
  auto gemm = [&](const unsigned short* A, const unsigned short* Bt, const float* bias,
                  const float* resF, float* outF, unsigned short* outH,
                  int M, int K, int N, int act) {
    int tilesN = N / 16;
    int total = (M / 32) * tilesN;  // 32 rows per wave
    gemm_wmma_kernel<<<(total + 7) / 8, 256, 0, stream>>>(
        A, Bt, bias, resF, outF, outH, M, K, N, act, tilesN, total);
  };

  // ---- backbone
  stem_kernel<<<gblk((long long)8 * 256 * 256 * 32), 256, 0, stream>>>(
      F(IN_X), F(IN_STEM_W), F(IN_STEM_B), A_stem);
  // stage 0: 32->64 @128
  conv(A_stem, w_s0d, F(IN_S0D_B), nullptr, S1A, 8, 256, 256, 32, 64, 2, 1);
  conv(S1A, w_s0b0, F(IN_S0B0_B), S1A, S1B, 8, 128, 128, 64, 64, 1, 2);
  unsigned short* F1 = S1B;
  // stage 1: 64->128 @64
  conv(F1, w_s1d, F(IN_S1D_B), nullptr, S2A, 8, 128, 128, 64, 128, 2, 1);
  conv(S2A, w_s1b0, F(IN_S1B0_B), S2A, S2B, 8, 64, 64, 128, 128, 1, 2);
  conv(S2B, w_s1b1, F(IN_S1B1_B), S2B, S2A, 8, 64, 64, 128, 128, 1, 2);
  unsigned short* F2 = S2A;
  // stage 2: 128->192 @32
  conv(F2, w_s2d, F(IN_S2D_B), nullptr, S3A, 8, 64, 64, 128, 192, 2, 1);
  conv(S3A, w_s2b0, F(IN_S2B0_B), S3A, S3B, 8, 32, 32, 192, 192, 1, 2);
  conv(S3B, w_s2b1, F(IN_S2B1_B), S3B, S3A, 8, 32, 32, 192, 192, 1, 2);
  conv(S3A, w_s2b2, F(IN_S2B2_B), S3A, S3B, 8, 32, 32, 192, 192, 1, 2);
  unsigned short* F3 = S3B;
  // stage 3: 192->256 @16
  conv(F3, w_s3d, F(IN_S3D_B), nullptr, S4A, 8, 32, 32, 192, 256, 2, 1);
  conv(S4A, w_s3b0, F(IN_S3B0_B), S4A, S4B, 8, 16, 16, 256, 256, 1, 2);
  conv(S4B, w_s3b1, F(IN_S3B1_B), S4B, S4A, 8, 16, 16, 256, 256, 1, 2);
  unsigned short* F4 = S4A;

  // ---- transformer blocks on p4/p5 (in place on feat buffer)
  auto run_tb = [&](unsigned short* feat, int C, int Ntok, int HD,
                    const float* ln1g, const float* ln1b, const float* ln2g,
                    const float* ln2b, unsigned short* qW, const float* qB,
                    unsigned short* pW, const float* pB, unsigned short* m1W,
                    const float* m1B, unsigned short* m2W, const float* m2B) {
    int M = 8 * Ntok;
    long long ne = (long long)M * C;
    cvt_h2f_kernel<<<gblk(ne), 256, 0, stream>>>(feat, Ttok, (int)ne);
    layernorm_kernel<<<gblk(M), 256, 0, stream>>>(Ttok, ln1g, ln1b, Th, M, C);
    gemm(Th, qW, qB, nullptr, nullptr, Tqkv, M, C, 3 * C, 0);
    int atot = 8 * 4 * Ntok;
    if (HD == 48)
      attn_kernel<48><<<gblk(atot), 256, 0, stream>>>(Tqkv, Tatt, 8, Ntok, C, 4);
    else
      attn_kernel<64><<<gblk(atot), 256, 0, stream>>>(Tqkv, Tatt, 8, Ntok, C, 4);
    gemm(Tatt, pW, pB, Ttok, Ttok, nullptr, M, C, C, 0);       // t += o@proj+b
    layernorm_kernel<<<gblk(M), 256, 0, stream>>>(Ttok, ln2g, ln2b, Th, M, C);
    gemm(Th, m1W, m1B, nullptr, nullptr, Tmlp, M, C, 4 * C, 2);  // gelu
    gemm(Tmlp, m2W, m2B, Ttok, Ttok, nullptr, M, 4 * C, C, 0);   // t += mlp
    cvt_f2h_kernel<<<gblk(ne), 256, 0, stream>>>(Ttok, feat, (int)ne);
  };
  run_tb(F3, 192, 1024, 48, F(IN_P4_LN1G), F(IN_P4_LN1B), F(IN_P4_LN2G),
         F(IN_P4_LN2B), w_q4, F(IN_P4_QB), w_p4, F(IN_P4_PB), w_m14,
         F(IN_P4_MB1), w_m24, F(IN_P4_MB2));
  run_tb(F4, 256, 256, 64, F(IN_P5_LN1G), F(IN_P5_LN1B), F(IN_P5_LN2G),
         F(IN_P5_LN2B), w_q5, F(IN_P5_QB), w_p5, F(IN_P5_PB), w_m15,
         F(IN_P5_MB1), w_m25, F(IN_P5_MB2));

  // ---- FPN neck: lateral 1x1 (GEMM) -> top-down add -> 3x3 smooth
  const unsigned short* feats[4] = {F1, F2, F3, F4};
  int lvlH[4] = {128, 64, 32, 16};
  int latC[4] = {64, 128, 192, 256};
  int latBi[4] = {IN_LAT2_B, IN_LAT3_B, IN_LAT4_B, IN_LAT5_B};
  int smBi[4]  = {IN_SM2_B, IN_SM3_B, IN_SM4_B, IN_SM5_B};
  for (int l = 0; l < 4; l++) {
    int M = 8 * lvlH[l] * lvlH[l];
    gemm(feats[l], w_lat[l], F(latBi[l]), nullptr, nullptr, LAT[l], M, latC[l], 128, 0);
  }
  for (int l = 2; l >= 0; l--) {  // p4 += up(p5); p3 += up(p4); p2 += up(p3)
    long long ne = (long long)8 * lvlH[l] * lvlH[l] * 128;
    add_up_kernel<<<gblk(ne), 256, 0, stream>>>(LAT[l], LAT[l + 1], 8, lvlH[l], lvlH[l], 128);
  }
  for (int l = 0; l < 4; l++)
    conv(LAT[l], w_sm[l], F(smBi[l]), nullptr, PYR[l], 8, lvlH[l], lvlH[l], 128, 128, 1, 0);

  // ---- heads
  int lvlOff[4] = {0, 16384, 20480, 21504};
  for (int l = 0; l < 4; l++) {
    int Hh = lvlH[l];
    conv(PYR[l], w_ct0, F(IN_CT0_B), nullptr, T1, 8, Hh, Hh, 128, 128, 1, 1);
    conv(T1, w_ct1, F(IN_CT1_B), nullptr, C2, 8, Hh, Hh, 128, 128, 1, 1);
    conv(PYR[l], w_rt0, F(IN_RT0_B), nullptr, T1, 8, Hh, Hh, 128, 128, 1, 1);
    conv(T1, w_rt1, F(IN_RT1_B), nullptr, R2, 8, Hh, Hh, 128, 128, 1, 1);
    long long nthr = (long long)8 * Hh * Hh * 15;
    head_final_kernel<<<gblk(nthr), 256, 0, stream>>>(
        C2, R2, F(IN_CLS_W), F(IN_CLS_B), F(IN_CTR_W), F(IN_CTR_B),
        F(IN_BOX_W), F(IN_BOX_B), (float*)d_out, 8, Hh, Hh, 128,
        lvlOff[l], 21760);
  }
}
